// Attention3D_824633721331
// MI455X (gfx1250) — matrix operand order repaired and checked by reading, not compiled, not run on hardware
//
#include <hip/hip_runtime.h>
#include <stdint.h>

typedef __attribute__((ext_vector_type(16))) _Float16 v16h;
typedef __attribute__((ext_vector_type(8)))  _Float16 v8h;
typedef __attribute__((ext_vector_type(8)))  float    v8f;
typedef uint32_t u32x4 __attribute__((ext_vector_type(4)));
typedef int      i32x8 __attribute__((ext_vector_type(8)));
typedef int      i32x4 __attribute__((ext_vector_type(4)));

#define N_TOK   4096
#define C_DIM   528
#define C_PAD   544      // 17*32, K-padded for x32 WMMA
#define NHEADS  8
#define HD      66
#define HD_PADQ 96       // 3*32 : K-pad for Q@K^T
#define HD_PADV 80       // 5*16 : N-pad for P@V
#define KCHUNKS 17       // C_PAD / 32
#define QSCALE  0.12309149097933274f  // 66^-0.5

// LDS row strides (f16) padded to avoid 4-way bank conflicts on b128 column reads
#define KROW 104         // 96 data + 8 pad  (208B -> stride 52 banks, 16 distinct)
#define VROW 40          // 32 data + 8 pad  (80B  -> stride 20 banks, 16 distinct)
#define PROW 40

// Tensor Data Mover availability (this toolchain: 6-arg builtin; else per-lane async)
#define HAVE_TDM 0
#if defined(__has_builtin)
#if __has_builtin(__builtin_amdgcn_tensor_load_to_lds)
#undef HAVE_TDM
#define HAVE_TDM 1
#endif
#endif

// ---- f16 workspace layout (element offsets) ----
#define OFF_XH  ((size_t)0)
#define OFF_WQ  (OFF_XH + (size_t)N_TOK * C_PAD)              // x f16, padded
#define OFF_WP  (OFF_WQ + (size_t)(3 * C_DIM) * C_PAD)        // w_qkv f16, padded
#define OFF_Q   (OFF_WP + (size_t)C_DIM * C_PAD)              // w_proj f16, padded
#define OFF_K   (OFF_Q  + (size_t)NHEADS * N_TOK * HD_PADQ)   // Q [h][tok][96] (pre-scaled)
#define OFF_V   (OFF_K  + (size_t)NHEADS * N_TOK * HD_PADQ)   // K [h][tok][96]
#define OFF_OC  (OFF_V  + (size_t)NHEADS * HD_PADV * N_TOK)   // Vt [h][hd][tok]
#define WS_F16_TOTAL (OFF_OC + (size_t)N_TOK * C_PAD)         // Oc [tok][544]

// ---------------- fragment loaders (wave32 layouts, CDNA5 ISA §7.12.2) ----
// A 16x32 f16: lane L: M=L%16; VGPR0-3 hold K=8*half+0..7, VGPR4-7 K=16+8*half+0..7
__device__ __forceinline__ v16h frag_a(const _Float16* p, int hf) {
  union { v16h v; v8h h[2]; } u;
  u.h[0] = *reinterpret_cast<const v8h*>(p + 8 * hf);
  u.h[1] = *reinterpret_cast<const v8h*>(p + 16 + 8 * hf);
  return u.v;
}
// B 32x16 f16: lane L: N=L%16 (caller picks row ptr for that column); K=8*half+0..7, 16+8*half+0..7
__device__ __forceinline__ v16h frag_b(const _Float16* p, int hf) {
  union { v16h v; v8h h[2]; } u;
  u.h[0] = *reinterpret_cast<const v8h*>(p + 8 * hf);
  u.h[1] = *reinterpret_cast<const v8h*>(p + 16 + 8 * hf);
  return u.v;
}
__device__ __forceinline__ v8f wmma16(v16h a, v16h b, v8f c) {
  return __builtin_amdgcn_wmma_f32_16x16x32_f16(false, a, false, b, (short)0, c,
                                                false, false);
}

#if HAVE_TDM
// One TDM descriptor: copy nrows rows of tile_w f16 from global (row pitch
// g_pitch elems) into LDS at lds_pitch elems, via iterate mode (08_async_tensor §8-9).
__device__ __forceinline__ void tdm_load_rows(uint32_t lds_addr, uint64_t gaddr,
                                              int tile_w, int g_pitch,
                                              int lds_pitch, int nrows) {
  u32x4 g0;
  g0[0] = 1u;                                                     // count=1 valid
  g0[1] = lds_addr;                                               // LDS byte addr
  g0[2] = (uint32_t)gaddr;                                        // global addr lo
  g0[3] = ((uint32_t)(gaddr >> 32) & 0x01FFFFFFu) | (2u << 30);   // addr hi | type=2
  i32x8 g1;
  g1[0] = (1 << 16) | (1 << 19);        // data_size=2B, iterate_enable=1
  g1[1] = 0;                            // abar=0 | tensor_dim0 lo16 (dim0=2^30)
  g1[2] = 0x4000;                       // tensor_dim0 hi16 | tensor_dim1 lo16 (2^16)
  g1[3] = 1 | (tile_w << 16);           // tensor_dim1 hi16 | tile_dim0
  g1[4] = 1;                            // tile_dim1=1 | tile_dim2=0
  g1[5] = g_pitch;                      // tensor_dim0_stride lo32 (elems)
  g1[6] = 0;
  g1[7] = 0;
  i32x4 g2;
  g2[0] = 1;                            // tensor_dim2 (unused in iterate mode)
  g2[1] = lds_pitch;                    // lds_addr_increment (elems)
  g2[2] = g_pitch;                      // global_addr_increment lo32 (elems)
  g2[3] = (nrows - 1) << 16;            // | iterate_count (0 => 1x)
  i32x4 g3 = {0, 0, 0, 0};
  i32x8 g4 = {0, 0, 0, 0, 0, 0, 0, 0};  // trailing group (6-arg toolchain form)
  __builtin_amdgcn_tensor_load_to_lds(g0, g1, g2, g3, g4, 0);
}
#else
// CDNA5 async copy fallback: 16B global -> LDS, tracked with ASYNCcnt
__device__ __forceinline__ void async_cp16(uint32_t lds_off, const _Float16* g) {
  asm volatile("global_load_async_to_lds_b128 %0, %1, off"
               :: "v"(lds_off), "v"((uint64_t)(uintptr_t)g)
               : "memory");
}
#endif

// ---------------- utility kernels ----------------
__global__ void k_zero(uint32_t* p, size_t ndw) {
  size_t i = (size_t)blockIdx.x * blockDim.x + threadIdx.x;
  size_t stride = (size_t)gridDim.x * blockDim.x;
  for (; i < ndw; i += stride) p[i] = 0u;
}

__global__ void k_cvt_pad(const float* __restrict__ src, _Float16* __restrict__ dst,
                          int rows, int cols, int dstride) {
  int idx = blockIdx.x * blockDim.x + threadIdx.x;
  if (idx >= rows * cols) return;
  int r = idx / cols, c = idx - r * cols;
  dst[(size_t)r * dstride + c] = (_Float16)src[idx];
}

// ---------------- QKV projection: [4096,544]x[1584,544]^T, fused scatter ----
__global__ __launch_bounds__(256) void k_qkv(const _Float16* __restrict__ xh,
                                             const _Float16* __restrict__ wq,
                                             const float* __restrict__ b_qkv,
                                             _Float16* __restrict__ Qo,
                                             _Float16* __restrict__ Ko,
                                             _Float16* __restrict__ Vt) {
  const int lane = threadIdx.x & 31;
  const int wave = threadIdx.x >> 5;
  const int tile = blockIdx.x * 8 + wave;
  if (tile >= 256 * 99) return;
  const int tm = tile / 99, tn = tile - 99 * tm;
  const int hf = lane >> 4, lx = lane & 15;

  const _Float16* arow = xh + (size_t)(tm * 16 + lx) * C_PAD;
  const _Float16* brow = wq + (size_t)(tn * 16 + lx) * C_PAD;
  v8f acc = 0.f;
#pragma unroll 4
  for (int kc = 0; kc < KCHUNKS; ++kc) {
    v16h a = frag_a(arow + kc * 32, hf);
    v16h b = frag_b(brow + kc * 32, hf);
    acc = wmma16(a, b, acc);
  }
  // C layout: reg r -> row M = r + 8*hf, col N = lx
  const int o = tn * 16 + lx;           // 0..1583, tiles never straddle 528/1056
  const float bias = b_qkv[o];
  if (o < C_DIM) {                      // Q (pre-scaled by hd^-0.5)
    int head = o / HD, d = o - head * HD;
    _Float16* qp = Qo + ((size_t)head * N_TOK + tm * 16 + 8 * hf) * HD_PADQ + d;
#pragma unroll
    for (int r = 0; r < 8; ++r) qp[(size_t)r * HD_PADQ] = (_Float16)((acc[r] + bias) * QSCALE);
  } else if (o < 2 * C_DIM) {           // K, row-major [tok][96]
    int oo = o - C_DIM;
    int head = oo / HD, d = oo - head * HD;
    _Float16* kp = Ko + ((size_t)head * N_TOK + tm * 16 + 8 * hf) * HD_PADQ + d;
#pragma unroll
    for (int r = 0; r < 8; ++r) kp[(size_t)r * HD_PADQ] = (_Float16)(acc[r] + bias);
  } else {                              // V, transposed [hd][tok]
    int oo = o - 2 * C_DIM;
    int head = oo / HD, d = oo - head * HD;
    _Float16* vp = Vt + ((size_t)head * HD_PADV + d) * N_TOK + tm * 16 + 8 * hf;
#pragma unroll
    for (int r = 0; r < 8; ++r) vp[r] = (_Float16)(acc[r] + bias);
  }
}

// ---- flash attention: block = 8 waves x 16 queries, K/V staged to LDS by TDM ----
__global__ __launch_bounds__(256) void k_attn(const _Float16* __restrict__ Qm,
                                              const _Float16* __restrict__ Km,
                                              const _Float16* __restrict__ Vt,
                                              _Float16* __restrict__ Oc) {
  __shared__ __align__(16) _Float16 kbuf[2][32 * KROW];   // 32 keys x 96(+8)
  __shared__ __align__(16) _Float16 vbuf[2][80 * VROW];   // 80 hd  x 32(+8)
  __shared__ __align__(16) _Float16 pbuf[8][16 * PROW];   // per-wave P tile

  const int tid  = threadIdx.x;
  const int lane = tid & 31;
  const int wave = tid >> 5;
  const int hf = lane >> 4, lx = lane & 15;
  const int head = blockIdx.y;
  const int qtile = blockIdx.x * 8 + wave;
  _Float16* pl = &pbuf[wave][0];

  const _Float16* Kb = Km + (size_t)head * N_TOK * HD_PADQ;
  const _Float16* Vb = Vt + (size_t)head * HD_PADV * N_TOK;

  const _Float16* Qrow = Qm + ((size_t)head * N_TOK + qtile * 16 + lx) * HD_PADQ;
  const v16h qf0 = frag_a(Qrow, hf);
  const v16h qf1 = frag_a(Qrow + 32, hf);
  const v16h qf2 = frag_a(Qrow + 64, hf);

  float m[8], l[8];
  v8f oacc[5];
#pragma unroll
  for (int r = 0; r < 8; ++r) { m[r] = -1e30f; l[r] = 0.f; }
#pragma unroll
  for (int t = 0; t < 5; ++t) oacc[t] = 0.f;

#if HAVE_TDM
  // wave 0 drives the Tensor Data Mover: one descriptor each for K and V tiles
  auto issue = [&](int buf, int kb) {
    tdm_load_rows((uint32_t)(uintptr_t)&kbuf[buf][0],
                  (uint64_t)(uintptr_t)(Kb + (size_t)kb * HD_PADQ),
                  HD_PADQ, HD_PADQ, KROW, 32);
    tdm_load_rows((uint32_t)(uintptr_t)&vbuf[buf][0],
                  (uint64_t)(uintptr_t)(Vb + kb),
                  32, N_TOK, VROW, HD_PADV);
  };
  if (wave == 0) issue(0, 0);
#else
  // cooperative per-lane async stage: K 192 + V 320 chunks = 2 instr/thread
  auto issue = [&](int buf, int kb) {
    uint32_t kbase = (uint32_t)(uintptr_t)&kbuf[buf][0];
    uint32_t vbase = (uint32_t)(uintptr_t)&vbuf[buf][0];
#pragma unroll
    for (int kIt = 0; kIt < 2; ++kIt) {
      int i = tid + kIt * 256;
      if (i < 192) {
        int row = i / 6, q = i - 6 * row;
        async_cp16(kbase + (uint32_t)(row * KROW + q * 8) * 2,
                   Kb + (size_t)(kb + row) * HD_PADQ + q * 8);
      } else {
        int j = i - 192;
        int row = j >> 2, q = j & 3;
        async_cp16(vbase + (uint32_t)(row * VROW + q * 8) * 2,
                   Vb + (size_t)row * N_TOK + kb + q * 8);
      }
    }
  };
  issue(0, 0);
#endif

  for (int it = 0; it < 128; ++it) {
    const int cur = it & 1;
    const int kb = it * 32;
#if HAVE_TDM
    if (wave == 0) {
      if (it + 1 < 128) {
        issue(cur ^ 1, kb + 32);                       // DMA next block
        __builtin_amdgcn_s_wait_tensorcnt((short)2);   // current block done
      } else {
        __builtin_amdgcn_s_wait_tensorcnt((short)0);
      }
    }
#else
    if (it + 1 < 128) {
      issue(cur ^ 1, kb + 32);
      asm volatile("s_wait_asynccnt 2" ::: "memory");
    } else {
      asm volatile("s_wait_asynccnt 0" ::: "memory");
    }
#endif
    __syncthreads();                                   // block's data landed

    // ---- S = Q @ K^T from LDS ----
    const _Float16* krow0 = &kbuf[cur][lx * KROW];          // keys kb+lx
    const _Float16* krow1 = &kbuf[cur][(16 + lx) * KROW];   // keys kb+16+lx
    v8f s0 = 0.f, s1 = 0.f;
    s0 = wmma16(qf0, frag_b(krow0, hf), s0);
    s0 = wmma16(qf1, frag_b(krow0 + 32, hf), s0);
    s0 = wmma16(qf2, frag_b(krow0 + 64, hf), s0);
    s1 = wmma16(qf0, frag_b(krow1, hf), s1);
    s1 = wmma16(qf1, frag_b(krow1 + 32, hf), s1);
    s1 = wmma16(qf2, frag_b(krow1 + 64, hf), s1);

    // ---- online softmax; row M = r + 8*hf lives in one 16-lane half ----
    float corr[8];
#pragma unroll
    for (int r = 0; r < 8; ++r) {
      float mx = fmaxf(s0[r], s1[r]);
      mx = fmaxf(mx, __shfl_xor(mx, 1, 32));
      mx = fmaxf(mx, __shfl_xor(mx, 2, 32));
      mx = fmaxf(mx, __shfl_xor(mx, 4, 32));
      mx = fmaxf(mx, __shfl_xor(mx, 8, 32));
      float mnew = fmaxf(m[r], mx);
      float cr = __expf(m[r] - mnew);
      corr[r] = cr;
      m[r] = mnew;
      float p0 = __expf(s0[r] - mnew);
      float p1 = __expf(s1[r] - mnew);
      pl[(r + 8 * hf) * PROW + lx] = (_Float16)p0;
      pl[(r + 8 * hf) * PROW + 16 + lx] = (_Float16)p1;
      float rs = p0 + p1;
      rs += __shfl_xor(rs, 1, 32);
      rs += __shfl_xor(rs, 2, 32);
      rs += __shfl_xor(rs, 4, 32);
      rs += __shfl_xor(rs, 8, 32);
      l[r] = l[r] * cr + rs;
    }
    asm volatile("s_wait_dscnt 0" ::: "memory");     // cross-lane P visibility

    union { v16h v; v8h h[2]; } pu;                  // P in A-fragment layout
    pu.h[0] = *reinterpret_cast<const v8h*>(pl + lx * PROW + 8 * hf);
    pu.h[1] = *reinterpret_cast<const v8h*>(pl + lx * PROW + 16 + 8 * hf);
    const v16h pf = pu.v;

    // ---- O += P @ V from LDS ----
#pragma unroll
    for (int t = 0; t < 5; ++t) {
      v16h vf = frag_b(&vbuf[cur][(t * 16 + lx) * VROW], hf);
      v8f oc = oacc[t];
#pragma unroll
      for (int r = 0; r < 8; ++r) oc[r] *= corr[r];
      oacc[t] = wmma16(pf, vf, oc);
    }
    __syncthreads();   // everyone done reading cur before it is refilled
  }

  // epilogue: O / l -> Oc[tok][head*66 + d]
#pragma unroll
  for (int t = 0; t < 5; ++t) {
    int d = t * 16 + lx;
    if (d < HD) {
      size_t col = (size_t)head * HD + d;
#pragma unroll
      for (int r = 0; r < 8; ++r) {
        size_t tok = (size_t)qtile * 16 + r + 8 * hf;
        Oc[tok * C_PAD + col] = (_Float16)(oacc[t][r] / l[r]);
      }
    }
  }
}

// ---------------- output projection: [4096,544]x[528,544]^T + bias -> f32 ----
__global__ __launch_bounds__(256) void k_proj(const _Float16* __restrict__ Oc,
                                              const _Float16* __restrict__ wp,
                                              const float* __restrict__ b_proj,
                                              float* __restrict__ out) {
  const int lane = threadIdx.x & 31;
  const int wave = threadIdx.x >> 5;
  const int tile = blockIdx.x * 8 + wave;
  if (tile >= 256 * 33) return;
  const int tm = tile / 33, tn = tile - 33 * tm;
  const int hf = lane >> 4, lx = lane & 15;

  const _Float16* arow = Oc + (size_t)(tm * 16 + lx) * C_PAD;
  const _Float16* brow = wp + (size_t)(tn * 16 + lx) * C_PAD;
  v8f acc = 0.f;
#pragma unroll 4
  for (int kc = 0; kc < KCHUNKS; ++kc) {
    v16h a = frag_a(arow + kc * 32, hf);
    v16h b = frag_b(brow + kc * 32, hf);
    acc = wmma16(a, b, acc);
  }
  const int col = tn * 16 + lx;
  const float bias = b_proj[col];
#pragma unroll
  for (int r = 0; r < 8; ++r) {
    size_t tok = (size_t)tm * 16 + r + 8 * hf;
    out[tok * C_DIM + col] = acc[r] + bias;
  }
}

// ---------------- launcher ----------------
extern "C" void kernel_launch(void* const* d_in, const int* in_sizes, int n_in,
                              void* d_out, int out_size, void* d_ws, size_t ws_size,
                              hipStream_t stream) {
  const float* x      = (const float*)d_in[0];
  const float* w_qkv  = (const float*)d_in[1];
  const float* b_qkv  = (const float*)d_in[2];
  const float* w_proj = (const float*)d_in[3];
  const float* b_proj = (const float*)d_in[4];
  float* out = (float*)d_out;

  _Float16* ws = (_Float16*)d_ws;
  _Float16* xh = ws + OFF_XH;
  _Float16* wq = ws + OFF_WQ;
  _Float16* wp = ws + OFF_WP;
  _Float16* Qm = ws + OFF_Q;
  _Float16* Km = ws + OFF_K;
  _Float16* Vt = ws + OFF_V;
  _Float16* Oc = ws + OFF_OC;

  // zero whole f16 workspace (establishes all zero-padding deterministically)
  k_zero<<<2048, 256, 0, stream>>>((uint32_t*)d_ws, WS_F16_TOTAL / 2);

  k_cvt_pad<<<(N_TOK * C_DIM + 255) / 256, 256, 0, stream>>>(x, xh, N_TOK, C_DIM, C_PAD);
  k_cvt_pad<<<(3 * C_DIM * C_DIM + 255) / 256, 256, 0, stream>>>(w_qkv, wq, 3 * C_DIM, C_DIM, C_PAD);
  k_cvt_pad<<<(C_DIM * C_DIM + 255) / 256, 256, 0, stream>>>(w_proj, wp, C_DIM, C_DIM, C_PAD);

  k_qkv<<<(256 * 99) / 8, 256, 0, stream>>>(xh, wq, b_qkv, Qm, Km, Vt);
  k_attn<<<dim3(32, NHEADS), 256, 0, stream>>>(Qm, Km, Vt, Oc);
  k_proj<<<(256 * 33 + 7) / 8, 256, 0, stream>>>(Oc, wp, b_proj, out);
}
